// encoder_43104291782771
// MI455X (gfx1250) — matrix operand order repaired and checked by reading, not compiled, not run on hardware
//
#include <hip/hip_runtime.h>
#include <math.h>

#define DEV __device__ __forceinline__

typedef __attribute__((ext_vector_type(16))) __bf16 v16bf;
typedef __attribute__((ext_vector_type(8)))  __bf16 v8bf;
typedef __attribute__((ext_vector_type(8)))  float  v8f;

constexpr int BQ  = 8;          // batch
constexpr int LQ  = 3072;       // sequence length
constexpr int DMQ = 512;        // model dim (== CF)
constexpr int MQ  = BQ * LQ;    // 24576 GEMM rows
constexpr int KTOP = 8;
constexpr int KHALF = 12;       // (25-1)/2 moving-average half window

constexpr int BM = 128, BN = 64, BK = 32;

DEV v16bf cat8(v8bf lo, v8bf hi) {
  return __builtin_shufflevector(lo, hi, 0, 1, 2, 3, 4, 5, 6, 7, 8, 9, 10, 11,
                                 12, 13, 14, 15);
}

// ---- WMMA fragment gathers per CDNA5 ISA 7.12.2 (wave32) ----
// A tile LDS layout: row-major [BM][BK]. Lane frag = row[8h+0..7] ++ row[8h+16..23]
// -> two 16B contiguous chunks -> 2x ds_load_b128.
DEV v16bf load_a_frag(const __bf16* As, int m0, int lane) {
  const __bf16* row = As + (m0 + (lane & 15)) * BK + 8 * (lane >> 4);
  v8bf lo = *(const v8bf*)(row);
  v8bf hi = *(const v8bf*)(row + 16);
  return cat8(lo, hi);
}
// B tile LDS layout: K-contiguous per column [BN][BK]. Lane frag = col[8h+0..7] ++ col[8h+16..23]
// -> two 16B contiguous chunks -> 2x ds_load_b128 (no scalarized u16 gathers).
DEV v16bf load_b_frag(const __bf16* Bs, int n0, int lane) {
  const __bf16* col = Bs + (n0 + (lane & 15)) * BK + 8 * (lane >> 4);
  v8bf lo = *(const v8bf*)(col);
  v8bf hi = *(const v8bf*)(col + 16);
  return cat8(lo, hi);
}

DEV v8f wmma_bf16(v16bf a, v16bf b, v8f c) {
  return __builtin_amdgcn_wmma_f32_16x16x32_bf16(false, a, false, b, (short)0, c,
                                                 false, false);
}

// ---- Generic GEMM: C[M,N] = op(A[M,K] @ W[K,N] + bias + resid) ----
template <bool HAS_BIAS, bool RESID, bool RELU>
__global__ __launch_bounds__(256) void gemm_bf16_wmma(
    const float* __restrict__ A, const float* __restrict__ W,
    const float* __restrict__ bias, const float* __restrict__ resid,
    float* __restrict__ C, int M, int N, int K) {
  __shared__ __align__(16) __bf16 As[BM * BK];  // row-major [BM][BK]
  __shared__ __align__(16) __bf16 Bs[BN * BK];  // K-contiguous [BN][BK]
  const int tid = threadIdx.x;
  const int m0 = blockIdx.y * BM;
  const int n0 = blockIdx.x * BN;
  const int wave = tid >> 5, lane = tid & 31;
  const int wm = (wave >> 1) * 32;  // 4 wave-rows
  const int wn = (wave & 1) * 32;   // 2 wave-cols
  v8f acc[2][2] = {};

  for (int k0 = 0; k0 < K; k0 += BK) {
    {  // stage A tile (fp32 -> bf16): thread handles 16 elems of one row-half
      const int r = tid >> 1, c = (tid & 1) * 16;
      const float* src = A + (size_t)(m0 + r) * K + k0 + c;
      __bf16* dst = As + r * BK + c;
#pragma unroll
      for (int j = 0; j < 16; j += 4) {
        float4 f = *(const float4*)(src + j);
        dst[j] = (__bf16)f.x; dst[j + 1] = (__bf16)f.y;
        dst[j + 2] = (__bf16)f.z; dst[j + 3] = (__bf16)f.w;
      }
    }
    {  // stage B tile transposed: col c gets W[k0+r][n0+c]; lanes coalesce on c
      const int c = tid & 63, r0 = (tid >> 6) * 8;
      const float* src = W + (size_t)(k0 + r0) * N + n0 + c;
      __bf16* dst = Bs + c * BK + r0;
#pragma unroll
      for (int j = 0; j < 8; ++j) dst[j] = (__bf16)src[(size_t)j * N];
    }
    __syncthreads();
    v16bf a0 = load_a_frag(As, wm, lane);
    v16bf a1 = load_a_frag(As, wm + 16, lane);
    v16bf b0 = load_b_frag(Bs, wn, lane);
    v16bf b1 = load_b_frag(Bs, wn + 16, lane);
    acc[0][0] = wmma_bf16(a0, b0, acc[0][0]);
    acc[0][1] = wmma_bf16(a0, b1, acc[0][1]);
    acc[1][0] = wmma_bf16(a1, b0, acc[1][0]);
    acc[1][1] = wmma_bf16(a1, b1, acc[1][1]);
    __syncthreads();
  }
  const int h = lane >> 4;
#pragma unroll
  for (int i = 0; i < 2; ++i)
#pragma unroll
    for (int j = 0; j < 2; ++j) {
      const int ncol = n0 + wn + 16 * j + (lane & 15);
      const float bv = HAS_BIAS ? bias[ncol] : 0.0f;
#pragma unroll
      for (int t = 0; t < 8; ++t) {
        const int mrow = m0 + wm + 16 * i + 8 * h + t;
        float val = acc[i][j][t] + bv;
        if (RESID) val += resid[(size_t)mrow * N + ncol];
        if (RELU) val = fmaxf(val, 0.0f);
        C[(size_t)mrow * N + ncol] = val;
      }
    }
}

// ---- Fused Q@K^T with circular-diagonal reduction -> mean_corr[B,L] ----
__global__ __launch_bounds__(256) void corr_diag_wmma(
    const float* __restrict__ Q, const float* __restrict__ Kmat,
    float* __restrict__ mean_corr) {
  __shared__ __align__(16) __bf16 As[BM * BK];
  __shared__ __align__(16) __bf16 Bs[BN * BK];   // K-contiguous per column
  __shared__ float bins[BM + BN - 1];            // 191 diagonals per tile
  const int tid = threadIdx.x;
  const int b = blockIdx.z;
  const int m0 = blockIdx.y * BM;
  const int n0 = blockIdx.x * BN;
  const float* Qb = Q + (size_t)b * LQ * DMQ;
  const float* Kb = Kmat + (size_t)b * LQ * DMQ;
  if (tid < BM + BN - 1) bins[tid] = 0.0f;
  const int wave = tid >> 5, lane = tid & 31;
  const int wm = (wave >> 1) * 32, wn = (wave & 1) * 32;
  v8f acc[2][2] = {};

  for (int k0 = 0; k0 < DMQ; k0 += BK) {
    {  // A = rows of Q
      const int r = tid >> 1, c = (tid & 1) * 16;
      const float* src = Qb + (size_t)(m0 + r) * DMQ + k0 + c;
      __bf16* dst = As + r * BK + c;
#pragma unroll
      for (int j = 0; j < 16; j += 4) {
        float4 f = *(const float4*)(src + j);
        dst[j] = (__bf16)f.x; dst[j + 1] = (__bf16)f.y;
        dst[j + 2] = (__bf16)f.z; dst[j + 3] = (__bf16)f.w;
      }
    }
    {  // B col c = K[n0+c][k0 .. k0+31]: contiguous reads AND contiguous writes
      const int c = tid & 63, r0 = (tid >> 6) * 8;
      const float* src = Kb + (size_t)(n0 + c) * DMQ + k0 + r0;
      __bf16* dst = Bs + c * BK + r0;
#pragma unroll
      for (int j = 0; j < 8; j += 4) {
        float4 f = *(const float4*)(src + j);
        dst[j] = (__bf16)f.x; dst[j + 1] = (__bf16)f.y;
        dst[j + 2] = (__bf16)f.z; dst[j + 3] = (__bf16)f.w;
      }
    }
    __syncthreads();
    v16bf a0 = load_a_frag(As, wm, lane);
    v16bf a1 = load_a_frag(As, wm + 16, lane);
    v16bf b0 = load_b_frag(Bs, wn, lane);
    v16bf b1 = load_b_frag(Bs, wn + 16, lane);
    acc[0][0] = wmma_bf16(a0, b0, acc[0][0]);
    acc[0][1] = wmma_bf16(a0, b1, acc[0][1]);
    acc[1][0] = wmma_bf16(a1, b0, acc[1][0]);
    acc[1][1] = wmma_bf16(a1, b1, acc[1][1]);
    __syncthreads();
  }
  // G[r,c] contributes to lag l = (r-c) mod L ; reduce per-tile into LDS bins
  const int h = lane >> 4;
#pragma unroll
  for (int i = 0; i < 2; ++i)
#pragma unroll
    for (int j = 0; j < 2; ++j) {
      const int cc = wn + 16 * j + (lane & 15);
#pragma unroll
      for (int t = 0; t < 8; ++t) {
        const int rr = wm + 16 * i + 8 * h + t;
        atomicAdd(&bins[rr - cc + (BN - 1)], acc[i][j][t]);
      }
    }
  __syncthreads();
  if (tid < BM + BN - 1) {
    const int diff = m0 - n0 + tid - (BN - 1);
    const int l = ((diff % LQ) + LQ) % LQ;
    atomicAdd(&mean_corr[b * LQ + l], bins[tid] * (1.0f / (float)DMQ));
  }
}

__global__ void zero_f32(float* p, int n) {
  int i = blockIdx.x * 256 + threadIdx.x;
  if (i < n) p[i] = 0.0f;
}

// ---- top-8 + softmax per batch ----
__global__ __launch_bounds__(256) void topk_softmax(
    const float* __restrict__ mean_corr, float* __restrict__ w,
    int* __restrict__ delays) {
  const int b = blockIdx.x;
  const int tid = threadIdx.x;
  __shared__ float vals[LQ];
  __shared__ float rmax[256];
  __shared__ int   ridx[256];
  __shared__ float topv[KTOP];
  __shared__ int   topi[KTOP];
  for (int i = tid; i < LQ; i += 256) vals[i] = mean_corr[b * LQ + i];
  __syncthreads();
  for (int it = 0; it < KTOP; ++it) {
    float best = -3.4e38f; int bi = 0;
    for (int i = tid; i < LQ; i += 256) {
      float v = vals[i];
      if (v > best) { best = v; bi = i; }
    }
    rmax[tid] = best; ridx[tid] = bi;
    __syncthreads();
    for (int s = 128; s > 0; s >>= 1) {
      if (tid < s && rmax[tid + s] > rmax[tid]) {
        rmax[tid] = rmax[tid + s]; ridx[tid] = ridx[tid + s];
      }
      __syncthreads();
    }
    if (tid == 0) {
      topv[it] = rmax[0]; topi[it] = ridx[0];
      vals[ridx[0]] = -3.4e38f;
    }
    __syncthreads();
  }
  if (tid == 0) {
    float mx = topv[0], s = 0.0f, e[KTOP];
    for (int i = 0; i < KTOP; ++i) { e[i] = __expf(topv[i] - mx); s += e[i]; }
    for (int i = 0; i < KTOP; ++i) {
      w[b * KTOP + i] = e[i] / s;
      delays[b * KTOP + i] = topi[i];
    }
  }
}

// ---- out[b,l,:] = sum_i w_i * v[b,(l+delay_i)%L,:] ----
__global__ __launch_bounds__(256) void roll_gather(
    const float* __restrict__ v, const float* __restrict__ w,
    const int* __restrict__ delays, float* __restrict__ out) {
  const int tid = blockIdx.x * 256 + threadIdx.x;
  const int D4 = DMQ / 4;
  const int d4 = tid & (D4 - 1);
  const int l = (tid >> 7) & (LQ - 1);
  const int b = tid / (LQ * D4);
  __shared__ float ws[KTOP];
  __shared__ int   ds[KTOP];
  if (threadIdx.x < KTOP) {
    ws[threadIdx.x] = w[b * KTOP + threadIdx.x];
    ds[threadIdx.x] = delays[b * KTOP + threadIdx.x];
  }
  __syncthreads();
  const float4* vb = (const float4*)(v + (size_t)b * LQ * DMQ);
  float4 acc = make_float4(0.f, 0.f, 0.f, 0.f);
#pragma unroll
  for (int i = 0; i < KTOP; ++i) {
    int src = l + ds[i];
    if (src >= LQ) src -= LQ;
    const float4 t = vb[src * D4 + d4];
    const float wi = ws[i];
    acc.x += wi * t.x; acc.y += wi * t.y; acc.z += wi * t.z; acc.w += wi * t.w;
  }
  ((float4*)out)[tid] = acc;
}

// ---- seasonal = x - movavg_25_same(x)  (count excludes padding) ----
__global__ __launch_bounds__(256) void decomp_seasonal(
    const float* __restrict__ x, float* __restrict__ seasonal) {
  const int tid = blockIdx.x * 256 + threadIdx.x;
  const int D4 = DMQ / 4;
  const int d4 = tid & (D4 - 1);
  const int l = (tid >> 7) & (LQ - 1);
  const int b = tid / (LQ * D4);
  const int lo = (l - KHALF > 0) ? (l - KHALF) : 0;
  const int hi = (l + KHALF < LQ - 1) ? (l + KHALF) : (LQ - 1);
  const float4* xb = (const float4*)(x + (size_t)b * LQ * DMQ);
  float4 s = make_float4(0.f, 0.f, 0.f, 0.f);
  for (int j = lo; j <= hi; ++j) {
    const float4 t = xb[j * D4 + d4];
    s.x += t.x; s.y += t.y; s.z += t.z; s.w += t.w;
  }
  const float inv = 1.0f / (float)(hi - lo + 1);
  const float4 c = xb[l * D4 + d4];
  float4 o;
  o.x = c.x - s.x * inv; o.y = c.y - s.y * inv;
  o.z = c.z - s.z * inv; o.w = c.w - s.w * inv;
  ((float4*)seasonal)[tid] = o;
}

extern "C" void kernel_launch(void* const* d_in, const int* in_sizes, int n_in,
                              void* d_out, int out_size, void* d_ws,
                              size_t ws_size, hipStream_t stream) {
  (void)in_sizes; (void)n_in; (void)out_size; (void)ws_size;
  const float* inputs = (const float*)d_in[0];
  const float* W_res = (const float*)d_in[1];
  const float* b_res = (const float*)d_in[2];
  const float* W_in = (const float*)d_in[3];
  const float* b_in = (const float*)d_in[4];
  const float* Wq = (const float*)d_in[5];
  const float* bq = (const float*)d_in[6];
  const float* Wk = (const float*)d_in[7];
  const float* bk = (const float*)d_in[8];
  const float* Wv = (const float*)d_in[9];
  const float* bv = (const float*)d_in[10];
  const float* Wo = (const float*)d_in[11];
  const float* bo = (const float*)d_in[12];
  const float* Wc1 = (const float*)d_in[13];
  const float* Wc2 = (const float*)d_in[14];

  const size_t MSZ = (size_t)MQ * DMQ;  // 12.58M floats per activation buffer
  float* proj = (float*)d_ws;
  float* x    = proj + MSZ;
  float* q    = x + MSZ;
  float* k    = q + MSZ;
  float* v    = k + MSZ;
  float* attn = v + MSZ;
  float* mean_corr = attn + MSZ;        // B*L
  float* wbuf = mean_corr + (size_t)BQ * LQ;  // B*8
  int*   delays = (int*)(wbuf + BQ * KTOP);   // B*8

  const dim3 blk(256);
  const dim3 gg(DMQ / BN, MQ / BM);                 // (8, 192)
  const dim3 gcorr(LQ / BN, LQ / BM, BQ);           // (48, 24, 8)
  const int nvec = BQ * LQ * (DMQ / 4);
  const int gvec = nvec / 256;                      // 12288

  // 1-5: projections
  gemm_bf16_wmma<true, false, false><<<gg, blk, 0, stream>>>(inputs, W_res, b_res, nullptr, proj, MQ, DMQ, DMQ);
  gemm_bf16_wmma<true, false, false><<<gg, blk, 0, stream>>>(proj, W_in, b_in, nullptr, x, MQ, DMQ, DMQ);
  gemm_bf16_wmma<true, false, false><<<gg, blk, 0, stream>>>(x, Wq, bq, nullptr, q, MQ, DMQ, DMQ);
  gemm_bf16_wmma<true, false, false><<<gg, blk, 0, stream>>>(x, Wk, bk, nullptr, k, MQ, DMQ, DMQ);
  gemm_bf16_wmma<true, false, false><<<gg, blk, 0, stream>>>(x, Wv, bv, nullptr, v, MQ, DMQ, DMQ);
  // 6: FFT autocorrelation == diagonal sums of Q@K^T (WMMA + fused reduction)
  zero_f32<<<(BQ * LQ + 255) / 256, blk, 0, stream>>>(mean_corr, BQ * LQ);
  corr_diag_wmma<<<gcorr, blk, 0, stream>>>(q, k, mean_corr);
  // 7: top-8 delays + softmax
  topk_softmax<<<BQ, blk, 0, stream>>>(mean_corr, wbuf, delays);
  // 8: weighted circular roll of V
  roll_gather<<<gvec, blk, 0, stream>>>(v, wbuf, delays, attn);
  // 9: output projection + residual (x2 -> reuse x)
  gemm_bf16_wmma<true, true, false><<<gg, blk, 0, stream>>>(attn, Wo, bo, proj, x, MQ, DMQ, DMQ);
  // 10: decomp (seasonal -> reuse proj)
  decomp_seasonal<<<gvec, blk, 0, stream>>>(x, proj);
  // 11-12: conv1x1 FFN (y1 -> reuse q, y2 -> reuse k)
  gemm_bf16_wmma<false, false, true><<<gg, blk, 0, stream>>>(proj, Wc1, nullptr, nullptr, q, MQ, DMQ, DMQ);
  gemm_bf16_wmma<false, false, false><<<gg, blk, 0, stream>>>(q, Wc2, nullptr, nullptr, k, MQ, DMQ, DMQ);
  // 13: final decomp -> d_out
  decomp_seasonal<<<gvec, blk, 0, stream>>>(k, (float*)d_out);
}
